// CorrelationModule_19018115187191
// MI455X (gfx1250) — matrix operand order repaired and checked by reading, not compiled, not run on hardware
//
#include <hip/hip_runtime.h>
#include <hip/hip_bf16.h>

// ---------------------------------------------------------------------------
// CorrelationModule for MI455X (gfx1250).
// NHWC activations (f16) + (tap,ic)-ordered weights so every im2col K-chunk is
// 32 contiguous f16 -> b128 loads, no divides, one bounds check per pixel/tap.
// Heavy 3x3 convs: implicit-GEMM on v_wmma_f32_16x16x32_f16, double-buffered
// LDS, 2 WMMA per wave per K-step, staging via gfx1250 async global->LDS DMA
// (ASYNCcnt) with a register-pipelined fallback.
// ---------------------------------------------------------------------------

typedef __attribute__((ext_vector_type(16))) _Float16 v16h;
typedef __attribute__((ext_vector_type(8)))  float    v8f;

#define BB 2
#define CC 128
#define HH 32
#define WW 64
#define DDISP 7
#define NIMG (BB * DDISP * DDISP)   // 98
#define HW   (HH * WW)              // 2048

// gfx1250 async global->LDS DMA path, if the toolchain declares the builtins.
#if defined(__HIP_DEVICE_COMPILE__)
# if __has_builtin(__builtin_amdgcn_global_load_async_to_lds_b128) && \
     __has_builtin(__builtin_amdgcn_s_wait_asynccnt)
#  define CORR_ASYNC_LDS 1
# endif
#endif
#ifndef CORR_ASYNC_LDS
# define CORR_ASYNC_LDS 0
#endif

#if CORR_ASYNC_LDS
// Builtin signature (from clang diagnostics): (as1 int4*, as3 int4*, int, int)
typedef int corr_v4i __attribute__((vector_size(16)));
typedef __attribute__((address_space(1))) corr_v4i GlobInt4;
typedef __attribute__((address_space(3))) corr_v4i LdsInt4;
#endif

union Frag  { v16h h; uint4 q[2]; unsigned u[8]; };
union Pack8 { _Float16 h[8]; uint4 q; };

// ---------------------------------------------------------------------------
// 1) Bilinear sample f2 at 49 displaced grids + concat with f1 -> X0 (f16)
//    X0 layout NHWC: [n=98][32][64][c=256], c<128 = f1, c>=128 = sampled f2.
// ---------------------------------------------------------------------------
__global__ __launch_bounds__(256)
void corr_build_input_kernel(const float* __restrict__ f1,
                             const float* __restrict__ f2,
                             const float* __restrict__ coords,
                             _Float16* __restrict__ X0)
{
    int idx = blockIdx.x * blockDim.x + threadIdx.x;
    if (idx >= NIMG * HW) return;
    int n = idx / HW;
    int p = idx - n * HW;
    int b = n / (DDISP * DDISP);
    int t = n - b * (DDISP * DDISP);
    int di = t / DDISP;           // first D axis -> x displacement d[i]
    int dj = t - di * DDISP;      // second D axis -> y displacement d[j]
    float dx = (float)(di - 3);
    float dy = (float)(dj - 3);

    const float* cb = coords + (size_t)b * 2 * HW;
    float sx = cb[p] + dx;        // channel 0 = x
    float sy = cb[HW + p] + dy;   // channel 1 = y

    float fx = floorf(sx), fy = floorf(sy);
    int x0 = (int)fx, y0 = (int)fy;
    float wx1 = sx - fx, wy1 = sy - fy;
    float wx0 = 1.0f - wx1, wy0 = 1.0f - wy1;

    bool vx0 = (x0 >= 0) && (x0 <= WW - 1);
    bool vx1 = (x0 + 1 >= 0) && (x0 + 1 <= WW - 1);
    bool vy0 = (y0 >= 0) && (y0 <= HH - 1);
    bool vy1 = (y0 + 1 >= 0) && (y0 + 1 <= HH - 1);
    int x0c = min(max(x0, 0), WW - 1);
    int x1c = min(max(x0 + 1, 0), WW - 1);
    int y0c = min(max(y0, 0), HH - 1);
    int y1c = min(max(y0 + 1, 0), HH - 1);

    float w00 = wx0 * wy0 * ((vx0 && vy0) ? 1.0f : 0.0f);
    float w10 = wx1 * wy0 * ((vx1 && vy0) ? 1.0f : 0.0f);
    float w01 = wx0 * wy1 * ((vx0 && vy1) ? 1.0f : 0.0f);
    float w11 = wx1 * wy1 * ((vx1 && vy1) ? 1.0f : 0.0f);

    int o00 = y0c * WW + x0c, o10 = y0c * WW + x1c;
    int o01 = y1c * WW + x0c, o11 = y1c * WW + x1c;

    const float* f1b = f1 + (size_t)b * CC * HW;
    const float* f2b = f2 + (size_t)b * CC * HW;
    _Float16* Xn = X0 + ((size_t)n * HW + p) * (2 * CC);   // NHWC row

    for (int c = 0; c < CC; ++c) {
        const float* pl = f2b + (size_t)c * HW;
        float v = w00 * pl[o00] + w10 * pl[o10] + w01 * pl[o01] + w11 * pl[o11];
        Xn[c]      = (_Float16)f1b[(size_t)c * HW + p];
        Xn[CC + c] = (_Float16)v;
    }
}

// ---------------------------------------------------------------------------
// 2) Weight reorder+cast: w [Cout][Cin][3][3] f32 -> Wk [Cout][tap][ic] f16
// ---------------------------------------------------------------------------
__global__ __launch_bounds__(256)
void corr_reorder_w_kernel(const float* __restrict__ w,
                           _Float16* __restrict__ Wk, int Cout, int Cin)
{
    int idx = blockIdx.x * blockDim.x + threadIdx.x;
    int K = Cin * 9;
    if (idx >= Cout * K) return;
    int oc = idx / K;
    int r  = idx - oc * K;
    int t  = r / Cin;
    int ic = r - t * Cin;
    Wk[idx] = (_Float16)w[((size_t)oc * Cin + ic) * 9 + t];
}

// ---------------------------------------------------------------------------
// 3) Implicit-GEMM 3x3 conv (pad=1), NHWC, v_wmma_f32_16x16x32_f16.
//    Block = 128 thr = 4 waves (2 wm x 2 wn). Block tile = 32oc x 64px.
//    Each wave: one 16x32 A fragment, two B fragments -> 2 WMMA per K-step.
//    Double-buffered LDS; staging via async global->LDS DMA (or register
//    pipeline) so load latency hides under WMMA.
// ---------------------------------------------------------------------------
__global__ __launch_bounds__(128)
void corr_conv3x3_wmma_kernel(const _Float16* __restrict__ X,   // [N][Hin][Win][Cin]
                              const _Float16* __restrict__ Wk,  // [Cout][9*Cin]
                              _Float16* __restrict__ Y,         // [N][Hout][Wout][Cout]
                              int N, int Cin, int Hin, int Win,
                              int Cout, int Hout, int Wout, int stride)
{
    (void)N;
    const int K = Cin * 9;
    const int P = Hout * Wout;
    const int mTiles = Cout >> 5;   // 32-oc tiles
    const int pTiles = P >> 6;      // 64-px tiles

    int bid = blockIdx.x;
    int n   = bid / (mTiles * pTiles);
    int r   = bid - n * (mTiles * pTiles);
    int mt  = r / pTiles;
    int pt  = r - mt * pTiles;

    const int tid  = threadIdx.x;
    const int lane = tid & 31;
    const int hi   = lane >> 4;
    const int wave = tid >> 5;
    const int wm   = wave >> 1;     // oc sub-tile (0..1)
    const int wn   = wave & 1;      // 32-px group (0..1)

    __shared__ __align__(16) _Float16 Atile[2][2][16][32];  // [buf][wm][row][k]
    __shared__ __align__(16) _Float16 Btile[2][64][40];     // [buf][px][k] (stride 40)

    const int ocBase = mt * 32;
    const int pxBase = pt * 64;
    const _Float16* Xn = X + (size_t)n * Hin * Win * Cin;

    // per-thread A staging coords: 8 contiguous f16 (one uint4)
    const int a_kk  = (tid * 8) & 31;
    const int a_row = ((tid * 8) >> 5) & 15;
    const int a_g   = (tid * 8) >> 9;
    const _Float16* a_src = Wk + (size_t)(ocBase + a_g * 16 + a_row) * K + a_kk;

    // per-thread B staging coords: 16 contiguous f16 (two uint4), one pixel
    const int b_px = tid >> 1;            // 0..63
    const int b_kk = (tid & 1) * 16;      // 0 or 16
    const int pg   = pxBase + b_px;
    const int b_oy = pg / Wout;
    const int b_ox = pg - b_oy * Wout;

    const int nIC = Cin >> 5;             // 32-wide ic chunks per tap
    const int S   = 9 * nIC;              // total K-steps

    // staged-step state (uniform across threads -> SGPRs)
    int st_kh = 0, st_kw = 0, st_icb = 0;

    uint4 ra = {0u, 0u, 0u, 0u}, rb0 = {0u, 0u, 0u, 0u}, rb1 = {0u, 0u, 0u, 0u};

    // issue: start global loads (into LDS via async DMA, or into registers)
    auto stage_issue = [&](int buf) {
        const int k0 = (st_kh * 3 + st_kw) * Cin + st_icb;
        const _Float16* asrc = a_src + k0;
        int iy = b_oy * stride + st_kh - 1;
        int ix = b_ox * stride + st_kw - 1;
        bool valid = (iy >= 0 && iy < Hin && ix >= 0 && ix < Win);
        const _Float16* bsrc =
            Xn + ((size_t)iy * Win + ix) * Cin + st_icb + b_kk;
#if CORR_ASYNC_LDS
        __builtin_amdgcn_global_load_async_to_lds_b128(
            (GlobInt4*)(void*)asrc,
            (LdsInt4*)(void*)&Atile[buf][a_g][a_row][a_kk], 0, 0);
        if (valid) {
            __builtin_amdgcn_global_load_async_to_lds_b128(
                (GlobInt4*)(void*)bsrc,
                (LdsInt4*)(void*)&Btile[buf][b_px][b_kk], 0, 0);
            __builtin_amdgcn_global_load_async_to_lds_b128(
                (GlobInt4*)(void*)(bsrc + 8),
                (LdsInt4*)(void*)&Btile[buf][b_px][b_kk + 8], 0, 0);
        } else {
            uint4 z = {0u, 0u, 0u, 0u};
            *(uint4*)(&Btile[buf][b_px][b_kk])     = z;
            *(uint4*)(&Btile[buf][b_px][b_kk + 8]) = z;
        }
#else
        (void)buf;
        __builtin_prefetch(asrc + 32, 0, 1);
        ra = *(const uint4*)asrc;
        if (valid) {
            rb0 = ((const uint4*)bsrc)[0];
            rb1 = ((const uint4*)bsrc)[1];
        } else {
            uint4 z = {0u, 0u, 0u, 0u};
            rb0 = z; rb1 = z;
        }
#endif
    };
    // commit: make the staged tile visible in LDS
    auto stage_commit = [&](int buf) {
#if CORR_ASYNC_LDS
        (void)buf;
        __builtin_amdgcn_s_wait_asynccnt(0);
#else
        *(uint4*)(&Atile[buf][a_g][a_row][a_kk]) = ra;
        *(uint4*)(&Btile[buf][b_px][b_kk])       = rb0;
        *(uint4*)(&Btile[buf][b_px][b_kk + 8])   = rb1;
#endif
    };
    auto advance = [&]() {
        st_icb += 32;
        if (st_icb == Cin) {
            st_icb = 0;
            if (++st_kw == 3) { st_kw = 0; ++st_kh; }
        }
    };

    v8f acc0 = {}, acc1 = {};

    stage_issue(0);
    advance();
    stage_commit(0);
    __syncthreads();

    for (int s = 0; s < S; ++s) {
        const int buf = s & 1;

        // ---- fragments (contiguous k order: element s of lane half hi = k 16*hi+s, A and B alike) ----
        Frag fa, fb0, fb1;
        {
            const uint4* ar = (const uint4*)(&Atile[buf][wm][lane & 15][0]);
            fa.q[0] = ar[hi * 2];       // lanes 0-15: K 0..7 ; lanes 16-31: K16..23
            fa.q[1] = ar[hi * 2 + 1];   // lanes 0-15: K 8..15; lanes 16-31: K24..31
            const uint4* b0 = (const uint4*)(&Btile[buf][wn * 32 + (lane & 15)][0]);
            fb0.q[0] = b0[hi * 2];
            fb0.q[1] = b0[hi * 2 + 1];
            const uint4* b1 = (const uint4*)(&Btile[buf][wn * 32 + 16 + (lane & 15)][0]);
            fb1.q[0] = b1[hi * 2];
            fb1.q[1] = b1[hi * 2 + 1];
        }

        const bool more = (s + 1 < S);
        if (more) { stage_issue(buf ^ 1); advance(); }   // loads in flight...

        acc0 = __builtin_amdgcn_wmma_f32_16x16x32_f16(
                   false, fa.h, false, fb0.h, (short)0, acc0, false, false);
        acc1 = __builtin_amdgcn_wmma_f32_16x16x32_f16(
                   false, fa.h, false, fb1.h, (short)0, acc1, false, false);

        if (more) stage_commit(buf ^ 1);                 // ...landed after WMMA

        __syncthreads();
    }

    // ---- store D (NHWC): 8 consecutive oc per lane -> one b128 store ----
    {
        const int ocOff = ocBase + wm * 16 + hi * 8;   // M = r + 8*hi
        _Float16* Yn = Y + (size_t)n * P * Cout + ocOff;

        Pack8 p0, p1;
        #pragma unroll
        for (int rr = 0; rr < 8; ++rr) p0.h[rr] = (_Float16)acc0[rr];
        #pragma unroll
        for (int rr = 0; rr < 8; ++rr) p1.h[rr] = (_Float16)acc1[rr];

        int px0 = pxBase + wn * 32 + (lane & 15);
        int px1 = px0 + 16;
        *(uint4*)(Yn + (size_t)px0 * Cout) = p0.q;
        *(uint4*)(Yn + (size_t)px1 * Cout) = p1.q;
    }
}

// ---------------------------------------------------------------------------
// 4) InstanceNorm (biased var) + ReLU, in place, NHWC.
//    grid = N * (C/32); block = 256 = 8 pixel-rows x 32 channels (coalesced).
// ---------------------------------------------------------------------------
__global__ __launch_bounds__(256)
void corr_instnorm_relu_kernel(_Float16* __restrict__ buf, int P, int C)
{
    const int CG = C >> 5;
    int n  = blockIdx.x / CG;
    int cg = blockIdx.x - n * CG;
    int ch = threadIdx.x & 31;
    int pr = threadIdx.x >> 5;

    _Float16* base = buf + (size_t)n * P * C + cg * 32 + ch;

    float s = 0.0f, q = 0.0f;
    for (int p = pr; p < P; p += 8) {
        float v = (float)base[(size_t)p * C];
        s += v; q += v * v;
    }
    __shared__ float ss[8][32], qq[8][32];
    __shared__ float mean_s[32], inv_s[32];
    ss[pr][ch] = s; qq[pr][ch] = q;
    __syncthreads();
    if (pr == 0) {
        float S = 0.0f, Q = 0.0f;
        #pragma unroll
        for (int i = 0; i < 8; ++i) { S += ss[i][ch]; Q += qq[i][ch]; }
        float m  = S / (float)P;
        float var = Q / (float)P - m * m;
        mean_s[ch] = m;
        inv_s[ch]  = rsqrtf(var + 1e-5f);
    }
    __syncthreads();
    float m = mean_s[ch], inv = inv_s[ch];
    for (int p = pr; p < P; p += 8) {
        float v = ((float)base[(size_t)p * C] - m) * inv;
        base[(size_t)p * C] = (_Float16)(v > 0.0f ? v : 0.0f);
    }
}

// ---------------------------------------------------------------------------
// 5) ConvTranspose2d 4x4 stride2 pad1, 64->32 (direct, ~1.6 GMAC)
//    A4 NHWC [98][16][32][64], w_dt [64][32][4][4], A5 NHWC [98][32][64][32]
// ---------------------------------------------------------------------------
__global__ __launch_bounds__(256)
void corr_convT_kernel(const _Float16* __restrict__ A4,
                       const float* __restrict__ w_dt,
                       _Float16* __restrict__ A5)
{
    int idx = blockIdx.x * blockDim.x + threadIdx.x;
    const int total = NIMG * HW * 32;
    if (idx >= total) return;
    int oc   = idx & 31;
    int rest = idx >> 5;
    int n    = rest / HW;
    int p    = rest - n * HW;
    int oy   = p / WW, ox = p - oy * WW;

    int kh0 = (oy + 1) & 1;
    int kw0 = (ox + 1) & 1;
    float acc = 0.0f;
    const _Float16* a_img = A4 + (size_t)n * 16 * 32 * 64;
    #pragma unroll
    for (int dkh = 0; dkh < 2; ++dkh) {
        int kh = kh0 + 2 * dkh;
        int num_h = oy + 1 - kh;
        if (num_h < 0) continue;
        int ih = num_h >> 1;
        if (ih >= 16) continue;
        #pragma unroll
        for (int dkw = 0; dkw < 2; ++dkw) {
            int kw = kw0 + 2 * dkw;
            int num_w = ox + 1 - kw;
            if (num_w < 0) continue;
            int iw = num_w >> 1;
            if (iw >= 32) continue;
            const _Float16* a = a_img + ((size_t)ih * 32 + iw) * 64;  // contiguous ic
            const float* wp = w_dt + ((size_t)oc * 4 + kh) * 4 + kw;  // + ic*512
            float sum = 0.0f;
            for (int ic = 0; ic < 64; ++ic)
                sum += (float)a[ic] * wp[(size_t)ic * 512];
            acc += sum;
        }
    }
    A5[((size_t)n * HW + p) * 32 + oc] = (_Float16)acc;
}

// ---------------------------------------------------------------------------
// 6) 3x3 conv 32->1 + bias -> cost f32 [98][2048]  (A5 NHWC)
// ---------------------------------------------------------------------------
__global__ __launch_bounds__(256)
void corr_conv_out_kernel(const _Float16* __restrict__ A5,
                          const float* __restrict__ w_out,
                          const float* __restrict__ b_out,
                          float* __restrict__ cost)
{
    int idx = blockIdx.x * blockDim.x + threadIdx.x;
    if (idx >= NIMG * HW) return;
    int n = idx / HW;
    int p = idx - n * HW;
    int y = p / WW, x = p - y * WW;
    float acc = b_out[0];
    const _Float16* An = A5 + (size_t)n * HW * 32;
    #pragma unroll
    for (int kh = 0; kh < 3; ++kh) {
        int iy = y + kh - 1;
        if (iy < 0 || iy >= HH) continue;
        #pragma unroll
        for (int kw = 0; kw < 3; ++kw) {
            int ix = x + kw - 1;
            if (ix < 0 || ix >= WW) continue;
            const _Float16* a = An + ((size_t)iy * WW + ix) * 32;  // contiguous ic
            const float* wr = w_out + kh * 3 + kw;                 // + ic*9
            float sum = 0.0f;
            for (int ic = 0; ic < 32; ++ic)
                sum += (float)a[ic] * wr[ic * 9];
            acc += sum;
        }
    }
    cost[idx] = acc;
}

// ---------------------------------------------------------------------------
// 7) DAP 49x49 1x1 mix -> final output [2][49][32][64] f32
// ---------------------------------------------------------------------------
__global__ __launch_bounds__(256)
void corr_dap_kernel(const float* __restrict__ cost,
                     const float* __restrict__ w_dap,
                     float* __restrict__ out)
{
    int idx = blockIdx.x * blockDim.x + threadIdx.x;
    const int total = BB * 49 * HW;
    if (idx >= total) return;
    int b = idx / (49 * HW);
    int r = idx - b * (49 * HW);
    int o = r / HW;
    int p = r - o * HW;
    const float* cb = cost + (size_t)b * 49 * HW + p;
    const float* wr = w_dap + o * 49;
    float acc = 0.0f;
    for (int i = 0; i < 49; ++i) acc += wr[i] * cb[(size_t)i * HW];
    out[idx] = acc;
}

// ---------------------------------------------------------------------------
// Launcher
// ---------------------------------------------------------------------------
static inline size_t alignup(size_t x) { return (x + 255) & ~(size_t)255; }

extern "C" void kernel_launch(void* const* d_in, const int* in_sizes, int n_in,
                              void* d_out, int out_size, void* d_ws, size_t ws_size,
                              hipStream_t stream)
{
    (void)in_sizes; (void)n_in; (void)out_size; (void)ws_size;
    const float* f1     = (const float*)d_in[0];
    const float* f2     = (const float*)d_in[1];
    const float* coords = (const float*)d_in[2];
    const float* w1     = (const float*)d_in[3];
    const float* w2     = (const float*)d_in[4];
    const float* w3     = (const float*)d_in[5];
    const float* w4     = (const float*)d_in[6];
    const float* w_dt   = (const float*)d_in[7];
    const float* w_out  = (const float*)d_in[8];
    const float* b_out  = (const float*)d_in[9];
    const float* w_dap  = (const float*)d_in[10];
    float* out = (float*)d_out;

    // workspace carve-up (all NHWC)
    char* ws = (char*)d_ws;
    size_t off = 0;
    _Float16* X0 = (_Float16*)(ws + off); off += alignup((size_t)NIMG * HW  * 256 * 2);
    _Float16* A1 = (_Float16*)(ws + off); off += alignup((size_t)NIMG * HW  * 96  * 2);
    _Float16* A2 = (_Float16*)(ws + off); off += alignup((size_t)NIMG * 512 * 128 * 2);
    _Float16* A3 = (_Float16*)(ws + off); off += alignup((size_t)NIMG * 512 * 128 * 2);
    _Float16* A4 = (_Float16*)(ws + off); off += alignup((size_t)NIMG * 512 * 64  * 2);
    _Float16* A5 = (_Float16*)(ws + off); off += alignup((size_t)NIMG * HW  * 32  * 2);
    float*  COST = (float*)(ws + off);    off += alignup((size_t)NIMG * HW * 4);
    _Float16* W1h = (_Float16*)(ws + off); off += alignup((size_t)96  * 2304 * 2);
    _Float16* W2h = (_Float16*)(ws + off); off += alignup((size_t)128 * 864  * 2);
    _Float16* W3h = (_Float16*)(ws + off); off += alignup((size_t)128 * 1152 * 2);
    _Float16* W4h = (_Float16*)(ws + off); off += alignup((size_t)64  * 1152 * 2);

    // weight reorder + cast to [Cout][tap][ic] f16
    {
        int n1 = 96 * 2304, n2 = 128 * 864, n3 = 128 * 1152, n4 = 64 * 1152;
        corr_reorder_w_kernel<<<(n1 + 255) / 256, 256, 0, stream>>>(w1, W1h, 96, 256);
        corr_reorder_w_kernel<<<(n2 + 255) / 256, 256, 0, stream>>>(w2, W2h, 128, 96);
        corr_reorder_w_kernel<<<(n3 + 255) / 256, 256, 0, stream>>>(w3, W3h, 128, 128);
        corr_reorder_w_kernel<<<(n4 + 255) / 256, 256, 0, stream>>>(w4, W4h, 64, 128);
    }

    // sample + concat
    {
        int n = NIMG * HW;
        corr_build_input_kernel<<<(n + 255) / 256, 256, 0, stream>>>(f1, f2, coords, X0);
    }

    // conv1: 256 -> 96, 32x64, stride 1
    corr_conv3x3_wmma_kernel<<<NIMG * (96 / 32) * (HW / 64), 128, 0, stream>>>(
        X0, W1h, A1, NIMG, 256, HH, WW, 96, HH, WW, 1);
    corr_instnorm_relu_kernel<<<NIMG * (96 / 32), 256, 0, stream>>>(A1, HW, 96);

    // conv2: 96 -> 128, stride 2 -> 16x32
    corr_conv3x3_wmma_kernel<<<NIMG * (128 / 32) * (512 / 64), 128, 0, stream>>>(
        A1, W2h, A2, NIMG, 96, HH, WW, 128, 16, 32, 2);
    corr_instnorm_relu_kernel<<<NIMG * (128 / 32), 256, 0, stream>>>(A2, 512, 128);

    // conv3: 128 -> 128, 16x32, stride 1
    corr_conv3x3_wmma_kernel<<<NIMG * (128 / 32) * (512 / 64), 128, 0, stream>>>(
        A2, W3h, A3, NIMG, 128, 16, 32, 128, 16, 32, 1);
    corr_instnorm_relu_kernel<<<NIMG * (128 / 32), 256, 0, stream>>>(A3, 512, 128);

    // conv4: 128 -> 64, 16x32, stride 1
    corr_conv3x3_wmma_kernel<<<NIMG * (64 / 32) * (512 / 64), 128, 0, stream>>>(
        A3, W4h, A4, NIMG, 128, 16, 32, 64, 16, 32, 1);
    corr_instnorm_relu_kernel<<<NIMG * (64 / 32), 256, 0, stream>>>(A4, 512, 64);

    // convT: 64 -> 32, up to 32x64
    {
        int n = NIMG * HW * 32;
        corr_convT_kernel<<<(n + 255) / 256, 256, 0, stream>>>(A4, w_dt, A5);
    }
    corr_instnorm_relu_kernel<<<NIMG * (32 / 32), 256, 0, stream>>>(A5, HW, 32);

    // conv_out + bias -> cost
    {
        int n = NIMG * HW;
        corr_conv_out_kernel<<<(n + 255) / 256, 256, 0, stream>>>(A5, w_out, b_out, COST);
    }

    // DAP -> final output
    {
        int n = BB * 49 * HW;
        corr_dap_kernel<<<(n + 255) / 256, 256, 0, stream>>>(COST, w_dap, out);
    }
}
